// EmbedMatcher_88441966559594
// MI455X (gfx1250) — matrix operand order repaired and checked by reading, not compiled, not run on hardware
//
#include <hip/hip_runtime.h>

// ---------------------------------------------------------------------------
// MI455X (gfx1250) implementation: wave32, WMMA f32<-f16 16x16x32 matrix cores,
// f16 embedding table + double-buffered async Global->LDS staging.
// ---------------------------------------------------------------------------

typedef __attribute__((ext_vector_type(16))) _Float16 v16h;
typedef __attribute__((ext_vector_type(8)))  float    v8f;
typedef __attribute__((__vector_size__(16))) int      v4i32;

#define B_SZ   1024
#define FEW    5
#define KN     128     // neighbors per node
#define DD     256     // D
#define DM     512
#define DI     1024
#define HID    1024
#define G4     4096    // 4*HID
#define NSYM   100000
#define STEPS  4
#define EPS_LN 0.001f
#define LEAK   0.01f
#define CK     64      // neighbor-kernel k-chunk (halves)

#if defined(__gfx1250__) && __has_builtin(__builtin_amdgcn_global_load_async_to_lds_b128) && __has_builtin(__builtin_amdgcn_s_wait_asynccnt)
#define ASYNC_LDS 1
#endif

// 16-byte Global->LDS copy: async (ASYNCcnt-tracked) on gfx1250.
__device__ __forceinline__ void copy16_g2l(const void* g, void* l) {
#ifdef ASYNC_LDS
    __builtin_amdgcn_global_load_async_to_lds_b128(
        (__attribute__((address_space(1))) v4i32*)(void*)g,
        (__attribute__((address_space(3))) v4i32*)l, 0, 0);
#else
    *reinterpret_cast<int4*>(l) = *reinterpret_cast<const int4*>(g);
#endif
}
__device__ __forceinline__ void copy_g2l_wait() {
#ifdef ASYNC_LDS
    __builtin_amdgcn_s_wait_asynccnt(0);
#endif
}

__device__ __forceinline__ float sigf(float x) { return 1.0f / (1.0f + __expf(-x)); }

// block-wide sum reduce, blockDim.x == 256, red must hold >= 8 floats
__device__ __forceinline__ float blockReduceSum256(float v, volatile float* red) {
    __syncthreads();
    for (int m = 16; m > 0; m >>= 1) v += __shfl_xor(v, m, 32);
    int w = threadIdx.x >> 5, l = threadIdx.x & 31;
    if (l == 0) red[w] = v;
    __syncthreads();
    if (threadIdx.x == 0) {
        float s = 0.f;
        for (int i = 0; i < 8; i++) s += red[i];
        red[0] = s;
    }
    __syncthreads();
    return red[0];
}

// ---------------------------------------------------------------------------
// f32 -> f16 elementwise (embedding table prep)
// ---------------------------------------------------------------------------
__global__ void cvt_f16_kernel(const float* __restrict__ src, _Float16* __restrict__ dst,
                               size_t n) {
    size_t i = (size_t)blockIdx.x * blockDim.x + threadIdx.x;
    if (i < n) dst[i] = (_Float16)src[i];
}

// ---------------------------------------------------------------------------
// Weight prep: Wt[n*Kd + k] = (f16) W[k*N + n]   (transpose + f32->f16)
// ---------------------------------------------------------------------------
__global__ void transpose_f16_kernel(const float* __restrict__ W,
                                     _Float16* __restrict__ Wt,
                                     int Kd, int N) {
    size_t idx = (size_t)blockIdx.x * blockDim.x + threadIdx.x;
    if (idx >= (size_t)Kd * N) return;
    int n = (int)(idx / Kd);
    int k = (int)(idx % Kd);
    Wt[idx] = (_Float16)W[(size_t)k * N + n];
}

// ---------------------------------------------------------------------------
// Fused neighbor encoder.  One block = one sample.
//   proj = leaky_relu([emb[rel]|emb[ent]] @ gcn_W + Wb + b)   (128 x 256)
//   w    = softmax(proj @ attn_W + attn_b)  over 128 neighbors
//   agg  = sum_k proj[k]*w[k]
//   gate = sigmoid(agg . gate_W + gate_Wb + gate_b)
//   out  = tanh(gate*agg + (1-gate)*emb[self])
// GEMM: 8 waves, wave w owns rows [16w,16w+16), full 256 cols in registers.
// A rows are gathered from the f16 embedding table with async Global->LDS
// copies, double-buffered so chunk c+1 streams in while chunk c computes.
// B fragments are software double-buffered so a load is always in flight
// under each WMMA.
// ---------------------------------------------------------------------------
__global__ __launch_bounds__(256) void neighbor_encoder_kernel(
    const int* __restrict__ conn,        // (N, 128, 2)
    const int* __restrict__ self_ids,    // strided
    int self_stride,
    const float* __restrict__ emb,       // (NSYM+1, 256) f32 (self embedding)
    const _Float16* __restrict__ embh,   // (NSYM+1, 256) f16 (gather source)
    const _Float16* __restrict__ gcnWt,  // (256, 512) f16, row = out col, contiguous K
    const float* __restrict__ gcn_Wb,
    const float* __restrict__ gcn_b,
    const float* __restrict__ attn_W,
    const float* __restrict__ attn_b,
    const float* __restrict__ gate_W,
    const float* __restrict__ gate_Wb,
    const float* __restrict__ gate_b,
    float* __restrict__ out)             // (N, 256)
{
    __shared__ _Float16 As[2][KN][CK + 8];  // 2 x 128 x 72 halves (~36KB), 16B-aligned rows
    __shared__ float sc[KN];
    __shared__ float sw[KN];
    __shared__ float agg[DD];
    __shared__ float red[8];
    __shared__ float gsh;

    const int n    = blockIdx.x;
    const int tid  = threadIdx.x;
    const int wave = tid >> 5;
    const int lane = tid & 31;
    const int lrow = lane & 15;
    const int lhi  = lane >> 4;

    const int* cn = conn + (size_t)n * KN * 2;

    v8f acc[16];
#pragma unroll
    for (int t = 0; t < 16; ++t)
#pragma unroll
        for (int e = 0; e < 8; ++e) acc[t][e] = 0.0f;

    // stage one 128x64 chunk of [rel|ent] (gathered rows) into As[buf]
    const int srow = tid >> 1;            // neighbor row 0..127
    const int scb  = (tid & 1) * 32;      // half-offset within chunk (32 halves = 64B)
    auto stage = [&](int chunk, int buf) {
        const int kk    = chunk * CK;
        const int which = (kk < DD) ? 0 : 1;   // 0: rel column, 1: ent column
        const int off   = kk & (DD - 1);
        const int sym   = cn[srow * 2 + which];
        const _Float16* srce = embh + (size_t)sym * DD + off + scb;
        _Float16* dste = &As[buf][srow][scb];
#pragma unroll
        for (int j = 0; j < 4; ++j)            // 4 x 16B = 64B per thread
            copy16_g2l(srce + j * 8, dste + j * 8);
    };

    stage(0, 0);
    copy_g2l_wait();
    __syncthreads();

    const int NCHUNK = (2 * DD) / CK;     // 8 chunks of K=64
    for (int c = 0; c < NCHUNK; ++c) {
        if (c + 1 < NCHUNK) stage(c + 1, (c + 1) & 1);   // async, overlaps compute
        const int kk  = c * CK;
        const int buf = c & 1;

#pragma unroll
        for (int kt = 0; kt < CK / 32; ++kt) {            // k=32 WMMA steps
            const int arow = wave * 16 + lrow;
            const _Float16* ap = &As[buf][arow][kt * 32 + lhi * 16];
            v16h a;
#pragma unroll
            for (int e = 0; e < 8; ++e) { a[e] = ap[e]; a[8 + e] = ap[8 + e]; }
            const int kglob = kk + kt * 32 + lhi * 16;
            const _Float16* wb = gcnWt + kglob;
            v16h bcur = *reinterpret_cast<const v16h*>(wb + (size_t)lrow * 512);
#pragma unroll
            for (int nt = 0; nt < 16; ++nt) {
                v16h bnext = bcur;
                if (nt < 15)
                    bnext = *reinterpret_cast<const v16h*>(
                        wb + (size_t)((nt + 1) * 16 + lrow) * 512);
                acc[nt] = __builtin_amdgcn_wmma_f32_16x16x32_f16(
                    false, a, false, bcur, (short)0, acc[nt], false, false);
                bcur = bnext;
            }
        }
        copy_g2l_wait();     // next chunk staged
        __syncthreads();
    }

    // bias + leaky relu + per-row attention-score partials
    float spart[8];
#pragma unroll
    for (int rr = 0; rr < 8; ++rr) spart[rr] = 0.0f;
#pragma unroll
    for (int nt = 0; nt < 16; ++nt) {
        const int col = nt * 16 + lrow;
        const float bv = gcn_Wb[col] + gcn_b[col];
        const float aw = attn_W[col];
#pragma unroll
        for (int rr = 0; rr < 8; ++rr) {
            float v = acc[nt][rr] + bv;
            v = (v > 0.0f) ? v : LEAK * v;
            acc[nt][rr] = v;
            spart[rr] += v * aw;
        }
    }
    // reduce score over the 16 lanes that share a row set (masks < 16 stay in half)
#pragma unroll
    for (int rr = 0; rr < 8; ++rr) {
        float s = spart[rr];
        for (int m = 8; m > 0; m >>= 1) s += __shfl_xor(s, m, 32);
        spart[rr] = s;
    }
    if (lrow == 0) {
#pragma unroll
        for (int rr = 0; rr < 8; ++rr)
            sc[wave * 16 + lhi * 8 + rr] = spart[rr] + attn_b[0];
    }
    __syncthreads();

    // softmax over 128 neighbors (wave 0), zero agg (everyone)
    if (wave == 0) {
        float mx = -1e30f;
        for (int i = lane; i < KN; i += 32) mx = fmaxf(mx, sc[i]);
        for (int m = 16; m > 0; m >>= 1) mx = fmaxf(mx, __shfl_xor(mx, m, 32));
        float s = 0.0f;
        for (int i = lane; i < KN; i += 32) { float e = __expf(sc[i] - mx); sw[i] = e; s += e; }
        for (int m = 16; m > 0; m >>= 1) s += __shfl_xor(s, m, 32);
        const float inv = 1.0f / s;
        for (int i = lane; i < KN; i += 32) sw[i] *= inv;
    }
    agg[tid] = 0.0f;
    __syncthreads();

    // weighted aggregation: each lane owns one column per n-tile
#pragma unroll
    for (int nt = 0; nt < 16; ++nt) {
        const int col = nt * 16 + lrow;
        float p = 0.0f;
#pragma unroll
        for (int rr = 0; rr < 8; ++rr)
            p += acc[nt][rr] * sw[wave * 16 + lhi * 8 + rr];
        atomicAdd(&agg[col], p);   // ds_add_f32
    }
    __syncthreads();

    // gate
    float gp = agg[tid] * gate_W[tid];
    float gsum = blockReduceSum256(gp, red);
    if (tid == 0) gsh = sigf(gsum + gate_Wb[0] + gate_b[0]);
    __syncthreads();

    const int   selfid = self_ids[(size_t)n * self_stride];
    const float se     = emb[(size_t)selfid * DD + tid];
    const float g      = gsh;
    out[(size_t)n * DD + tid] = tanhf(g * agg[tid] + (1.0f - g) * se);
}

// ---------------------------------------------------------------------------
// Generic WMMA GEMM: C[M,N] = act(A[M,Kd] @ W[Kd,N] + bias), W given as f16
// transposed Wt[N,Kd].  Block tile 128x64, k-chunk 128, 8 waves.
// act: 0 = none, 1 = relu
// ---------------------------------------------------------------------------
__global__ __launch_bounds__(256) void gemm_wmma_kernel(
    const float* __restrict__ A,
    const _Float16* __restrict__ Wt,
    const float* __restrict__ bias,
    float* __restrict__ C,
    int M, int N, int Kd, int act)
{
    __shared__ _Float16 As[128][136];

    const int m0   = blockIdx.x * 128;
    const int n0   = blockIdx.y * 64;
    const int tid  = threadIdx.x;
    const int wave = tid >> 5;
    const int lane = tid & 31;
    const int lrow = lane & 15;
    const int lhi  = lane >> 4;

    v8f acc[4];
#pragma unroll
    for (int t = 0; t < 4; ++t)
#pragma unroll
        for (int e = 0; e < 8; ++e) acc[t][e] = 0.0f;

    for (int kk = 0; kk < Kd; kk += 128) {
        const int r    = tid >> 1;
        const int cb   = (tid & 1) * 64;
        const int grow = m0 + r;
        const bool valid = (grow < M);
        const float* src = A + (size_t)grow * Kd + kk + cb;
#pragma unroll
        for (int j = 0; j < 16; ++j) {
            float4 v;
            if (valid) v = *reinterpret_cast<const float4*>(src + j * 4);
            else { v.x = 0.f; v.y = 0.f; v.z = 0.f; v.w = 0.f; }
            As[r][cb + j * 4 + 0] = (_Float16)v.x;
            As[r][cb + j * 4 + 1] = (_Float16)v.y;
            As[r][cb + j * 4 + 2] = (_Float16)v.z;
            As[r][cb + j * 4 + 3] = (_Float16)v.w;
        }
        __syncthreads();

#pragma unroll
        for (int kt = 0; kt < 4; ++kt) {
            const int arow = wave * 16 + lrow;
            const _Float16* ap = &As[arow][kt * 32 + lhi * 16];
            v16h a;
#pragma unroll
            for (int e = 0; e < 8; ++e) { a[e] = ap[e]; a[8 + e] = ap[8 + e]; }
            const int kglob = kk + kt * 32 + lhi * 16;
            const _Float16* wb = Wt + kglob;
            v16h bcur = *reinterpret_cast<const v16h*>(wb + (size_t)(n0 + lrow) * Kd);
#pragma unroll
            for (int nt = 0; nt < 4; ++nt) {
                v16h bnext = bcur;
                if (nt < 3)
                    bnext = *reinterpret_cast<const v16h*>(
                        wb + (size_t)(n0 + (nt + 1) * 16 + lrow) * Kd);
                acc[nt] = __builtin_amdgcn_wmma_f32_16x16x32_f16(
                    false, a, false, bcur, (short)0, acc[nt], false, false);
                bcur = bnext;
            }
        }
        __syncthreads();
    }

#pragma unroll
    for (int nt = 0; nt < 4; ++nt) {
        const int col = n0 + nt * 16 + lrow;
        const float bv = bias ? bias[col] : 0.0f;
#pragma unroll
        for (int rr = 0; rr < 8; ++rr) {
            const int row = m0 + wave * 16 + lhi * 8 + rr;
            if (row < M) {
                float v = acc[nt][rr] + bv;
                if (act == 1) v = (v > 0.0f) ? v : 0.0f;
                C[(size_t)row * N + col] = v;
            }
        }
    }
}

// ---------------------------------------------------------------------------
// combine: vec[b, 0:256] = (l1+l2)/2 ; vec[b, 256:512] = (r1+r2)/2
// ---------------------------------------------------------------------------
__global__ void combine_kernel(const float* __restrict__ l1, const float* __restrict__ l2,
                               const float* __restrict__ r1, const float* __restrict__ r2,
                               float* __restrict__ out, int Nrows) {
    int idx = blockIdx.x * blockDim.x + threadIdx.x;
    if (idx >= Nrows * DM) return;
    int b = idx / DM, d = idx % DM;
    float v = (d < DD) ? 0.5f * (l1[b * DD + d] + l2[b * DD + d])
                       : 0.5f * (r1[b * DD + d - DD] + r2[b * DD + d - DD]);
    out[idx] = v;
}

// ---------------------------------------------------------------------------
// layernorm with residual: z = y2 + x; LN over 512 with ddof=1
// ---------------------------------------------------------------------------
__global__ __launch_bounds__(256) void ln_residual_kernel(
    const float* __restrict__ x, const float* __restrict__ y2,
    const float* __restrict__ g, const float* __restrict__ bb,
    float* __restrict__ out) {
    __shared__ float red[8];
    const int row = blockIdx.x;
    const int tid = threadIdx.x;
    const size_t base = (size_t)row * DM;
    float z0 = y2[base + tid] + x[base + tid];
    float z1 = y2[base + tid + 256] + x[base + tid + 256];
    float mu = blockReduceSum256(z0 + z1, red) * (1.0f / DM);
    float d0 = z0 - mu, d1 = z1 - mu;
    float var = blockReduceSum256(d0 * d0 + d1 * d1, red) * (1.0f / (DM - 1));
    float inv = 1.0f / (sqrtf(var) + EPS_LN);
    out[base + tid]       = d0 * inv * g[tid]       + bb[tid];
    out[base + tid + 256] = d1 * inv * g[tid + 256] + bb[tid + 256];
}

// support_g = mean over FEW rows
__global__ void mean_kernel(const float* __restrict__ enc, float* __restrict__ sg) {
    int j = blockIdx.x * blockDim.x + threadIdx.x;
    if (j >= DM) return;
    float s = 0.f;
    for (int f = 0; f < FEW; ++f) s += enc[f * DM + j];
    sg[j] = s * (1.0f / FEW);
}

// bhh_eff[j] = b_hh[j] + sum_k support_g[k] * W_hh[512+k][j]
__global__ void sgc_kernel(const float* __restrict__ sg, const float* __restrict__ W_hh,
                           const float* __restrict__ b_hh, float* __restrict__ bhh_eff) {
    int j = blockIdx.x * blockDim.x + threadIdx.x;
    if (j >= G4) return;
    float s = b_hh[j];
    for (int k = 0; k < DM; ++k) s += sg[k] * W_hh[(size_t)(DM + k) * G4 + j];
    bhh_eff[j] = s;
}

// LSTM elementwise step.  hw==null on step 0 (h_r == 0): gates = x_proj + b_hh.
__global__ void lstm_ew_kernel(const float* __restrict__ xp, const float* __restrict__ hw,
                               const float* __restrict__ b_hh, const float* __restrict__ q,
                               float* __restrict__ c, float* __restrict__ h, int first) {
    int idx = blockIdx.x * blockDim.x + threadIdx.x;
    if (idx >= B_SZ * HID) return;
    int b = idx >> 10, j = idx & (HID - 1);
    const size_t gb = (size_t)b * G4;
    float gi, gf, gg, go;
    if (hw) {
        gi = xp[gb + j]            + hw[gb + j];
        gf = xp[gb + HID + j]      + hw[gb + HID + j];
        gg = xp[gb + 2 * HID + j]  + hw[gb + 2 * HID + j];
        go = xp[gb + 3 * HID + j]  + hw[gb + 3 * HID + j];
    } else {
        gi = xp[gb + j]            + b_hh[j];
        gf = xp[gb + HID + j]      + b_hh[HID + j];
        gg = xp[gb + 2 * HID + j]  + b_hh[2 * HID + j];
        go = xp[gb + 3 * HID + j]  + b_hh[3 * HID + j];
    }
    float cp = first ? 0.0f : c[idx];
    float cn = sigf(gf) * cp + sigf(gi) * tanhf(gg);
    c[idx] = cn;
    float hc = sigf(go) * tanhf(cn);
    if (j < DM) h[(size_t)b * DM + j] = q[(size_t)b * DM + j] + hc;
}

// out[b] = sum_j h[b][j] * sg[j]
__global__ __launch_bounds__(256) void dot_kernel(const float* __restrict__ h,
                                                  const float* __restrict__ sg,
                                                  float* __restrict__ out) {
    __shared__ float red[8];
    const int b = blockIdx.x;
    const int tid = threadIdx.x;
    float p = h[(size_t)b * DM + tid] * sg[tid] + h[(size_t)b * DM + tid + 256] * sg[tid + 256];
    float s = blockReduceSum256(p, red);
    if (tid == 0) out[b] = s;
}

// ---------------------------------------------------------------------------
extern "C" void kernel_launch(void* const* d_in, const int* in_sizes, int n_in,
                              void* d_out, int out_size, void* d_ws, size_t ws_size,
                              hipStream_t stream) {
    (void)in_sizes; (void)n_in; (void)out_size; (void)ws_size;

    const int*   query   = (const int*)d_in[0];
    const int*   support = (const int*)d_in[1];
    const int*   q_l1    = (const int*)d_in[2];
    const int*   q_l2    = (const int*)d_in[3];
    const int*   q_r1    = (const int*)d_in[5];
    const int*   q_r2    = (const int*)d_in[6];
    const int*   s_l1    = (const int*)d_in[8];
    const int*   s_l2    = (const int*)d_in[9];
    const int*   s_r1    = (const int*)d_in[11];
    const int*   s_r2    = (const int*)d_in[12];
    const float* emb     = (const float*)d_in[14];
    const float* gcn_W   = (const float*)d_in[15];
    const float* gcn_Wb  = (const float*)d_in[16];
    const float* gcn_b   = (const float*)d_in[17];
    const float* attn_W  = (const float*)d_in[18];
    const float* attn_b  = (const float*)d_in[19];
    const float* gate_W  = (const float*)d_in[20];
    const float* gate_Wb = (const float*)d_in[21];
    const float* gate_b  = (const float*)d_in[22];
    const float* p1_W    = (const float*)d_in[23];
    const float* p1_b    = (const float*)d_in[24];
    const float* p2_W    = (const float*)d_in[25];
    const float* p2_b    = (const float*)d_in[26];
    const float* ln_g    = (const float*)d_in[27];
    const float* ln_b    = (const float*)d_in[28];
    const float* W_ih    = (const float*)d_in[29];
    const float* W_hh    = (const float*)d_in[30];
    const float* b_ih    = (const float*)d_in[31];
    const float* b_hh    = (const float*)d_in[32];

    size_t off = 0;
    auto alloc = [&](size_t bytes) -> void* {
        void* p = (char*)d_ws + off;
        off += (bytes + 255) & ~(size_t)255;
        return p;
    };

    _Float16* embh  = (_Float16*)alloc((size_t)(NSYM + 1) * DD * 2); // f16 emb table
    _Float16* gcnWt = (_Float16*)alloc((size_t)DD * DM * 2);         // 256x512
    _Float16* p1Wt  = (_Float16*)alloc((size_t)DI * DM * 2);         // 1024x512
    _Float16* p2Wt  = (_Float16*)alloc((size_t)DM * DI * 2);         // 512x1024
    _Float16* WihT  = (_Float16*)alloc((size_t)G4 * DM * 2);         // 4096x512
    _Float16* WhhT  = (_Float16*)alloc((size_t)G4 * DM * 2);         // 4096x512 (top half of W_hh)

    float* tq0 = (float*)alloc((size_t)B_SZ * DD * 4);
    float* tq1 = (float*)alloc((size_t)B_SZ * DD * 4);
    float* tq2 = (float*)alloc((size_t)B_SZ * DD * 4);
    float* tq3 = (float*)alloc((size_t)B_SZ * DD * 4);
    float* ts0 = (float*)alloc((size_t)FEW * DD * 4);
    float* ts1 = (float*)alloc((size_t)FEW * DD * 4);
    float* ts2 = (float*)alloc((size_t)FEW * DD * 4);
    float* ts3 = (float*)alloc((size_t)FEW * DD * 4);

    float* qvec  = (float*)alloc((size_t)B_SZ * DM * 4);
    float* svec  = (float*)alloc((size_t)FEW * DM * 4);
    float* h1    = (float*)alloc((size_t)B_SZ * DI * 4);
    float* seout = (float*)alloc((size_t)B_SZ * DM * 4);
    float* senc  = (float*)alloc((size_t)FEW * DM * 4);
    float* qenc  = (float*)alloc((size_t)B_SZ * DM * 4);
    float* sg    = (float*)alloc((size_t)DM * 4);
    float* bhh   = (float*)alloc((size_t)G4 * 4);
    float* xproj = (float*)alloc((size_t)B_SZ * G4 * 4);
    float* hw    = (float*)alloc((size_t)B_SZ * G4 * 4);
    float* cbuf  = (float*)alloc((size_t)B_SZ * HID * 4);
    float* hbuf  = (float*)alloc((size_t)B_SZ * DM * 4);

    auto tgrid = [](size_t n) { return dim3((unsigned)((n + 255) / 256)); };

    // ---- prep: f16 embedding table + transposed f16 weights ----
    cvt_f16_kernel<<<tgrid((size_t)(NSYM + 1) * DD), 256, 0, stream>>>(
        emb, embh, (size_t)(NSYM + 1) * DD);
    transpose_f16_kernel<<<tgrid((size_t)DM * DD), 256, 0, stream>>>(gcn_W, gcnWt, DM, DD);
    transpose_f16_kernel<<<tgrid((size_t)DM * DI), 256, 0, stream>>>(p1_W, p1Wt, DM, DI);
    transpose_f16_kernel<<<tgrid((size_t)DI * DM), 256, 0, stream>>>(p2_W, p2Wt, DI, DM);
    transpose_f16_kernel<<<tgrid((size_t)DM * G4), 256, 0, stream>>>(W_ih, WihT, DM, G4);
    transpose_f16_kernel<<<tgrid((size_t)DM * G4), 256, 0, stream>>>(W_hh, WhhT, DM, G4);

    // ---- neighbor encoders (8 calls) ----
    neighbor_encoder_kernel<<<B_SZ, 256, 0, stream>>>(q_l1, query,     2, emb, embh, gcnWt,
        gcn_Wb, gcn_b, attn_W, attn_b, gate_W, gate_Wb, gate_b, tq0);
    neighbor_encoder_kernel<<<B_SZ, 256, 0, stream>>>(q_l2, query,     2, emb, embh, gcnWt,
        gcn_Wb, gcn_b, attn_W, attn_b, gate_W, gate_Wb, gate_b, tq1);
    neighbor_encoder_kernel<<<B_SZ, 256, 0, stream>>>(q_r1, query + 1, 2, emb, embh, gcnWt,
        gcn_Wb, gcn_b, attn_W, attn_b, gate_W, gate_Wb, gate_b, tq2);
    neighbor_encoder_kernel<<<B_SZ, 256, 0, stream>>>(q_r2, query + 1, 2, emb, embh, gcnWt,
        gcn_Wb, gcn_b, attn_W, attn_b, gate_W, gate_Wb, gate_b, tq3);
    neighbor_encoder_kernel<<<FEW, 256, 0, stream>>>(s_l1, support,     2, emb, embh, gcnWt,
        gcn_Wb, gcn_b, attn_W, attn_b, gate_W, gate_Wb, gate_b, ts0);
    neighbor_encoder_kernel<<<FEW, 256, 0, stream>>>(s_l2, support,     2, emb, embh, gcnWt,
        gcn_Wb, gcn_b, attn_W, attn_b, gate_W, gate_Wb, gate_b, ts1);
    neighbor_encoder_kernel<<<FEW, 256, 0, stream>>>(s_r1, support + 1, 2, emb, embh, gcnWt,
        gcn_Wb, gcn_b, attn_W, attn_b, gate_W, gate_Wb, gate_b, ts2);
    neighbor_encoder_kernel<<<FEW, 256, 0, stream>>>(s_r2, support + 1, 2, emb, embh, gcnWt,
        gcn_Wb, gcn_b, attn_W, attn_b, gate_W, gate_Wb, gate_b, ts3);

    combine_kernel<<<tgrid((size_t)B_SZ * DM), 256, 0, stream>>>(tq0, tq1, tq2, tq3, qvec, B_SZ);
    combine_kernel<<<tgrid((size_t)FEW * DM), 256, 0, stream>>>(ts0, ts1, ts2, ts3, svec, FEW);

    // ---- support encoder on support_vec (M=5) ----
    gemm_wmma_kernel<<<dim3(1, DI / 64), 256, 0, stream>>>(svec, p1Wt, p1_b, h1, FEW, DI, DM, 1);
    gemm_wmma_kernel<<<dim3(1, DM / 64), 256, 0, stream>>>(h1, p2Wt, p2_b, seout, FEW, DM, DI, 0);
    ln_residual_kernel<<<FEW, 256, 0, stream>>>(svec, seout, ln_g, ln_b, senc);
    mean_kernel<<<2, 256, 0, stream>>>(senc, sg);

    // ---- support encoder on query_vec (M=1024) ----
    gemm_wmma_kernel<<<dim3(B_SZ / 128, DI / 64), 256, 0, stream>>>(qvec, p1Wt, p1_b, h1, B_SZ, DI, DM, 1);
    gemm_wmma_kernel<<<dim3(B_SZ / 128, DM / 64), 256, 0, stream>>>(h1, p2Wt, p2_b, seout, B_SZ, DM, DI, 0);
    ln_residual_kernel<<<B_SZ, 256, 0, stream>>>(qvec, seout, ln_g, ln_b, qenc);

    // ---- LSTM query encoder ----
    // x_proj = qenc @ W_ih + b_ih
    gemm_wmma_kernel<<<dim3(B_SZ / 128, G4 / 64), 256, 0, stream>>>(qenc, WihT, b_ih, xproj, B_SZ, G4, DM, 0);
    // constant term: bhh = b_hh + support_g @ W_hh[512:]   (softmax over 1 col == 1 => r == support_g)
    sgc_kernel<<<G4 / 256, 256, 0, stream>>>(sg, W_hh, b_hh, bhh);

    // step 0: h_r == 0
    lstm_ew_kernel<<<tgrid((size_t)B_SZ * HID), 256, 0, stream>>>(xproj, nullptr, b_hh, qenc, cbuf, hbuf, 1);
    // steps 1..3: gates = x_proj + h @ W_hh[:512] + bhh
    for (int s = 1; s < STEPS; ++s) {
        gemm_wmma_kernel<<<dim3(B_SZ / 128, G4 / 64), 256, 0, stream>>>(hbuf, WhhT, bhh, hw, B_SZ, G4, DM, 0);
        lstm_ew_kernel<<<tgrid((size_t)B_SZ * HID), 256, 0, stream>>>(xproj, hw, b_hh, qenc, cbuf, hbuf, 0);
    }

    // ---- final scores ----
    dot_kernel<<<B_SZ, 256, 0, stream>>>(hbuf, sg, (float*)d_out);
}
